// DisNet_84104049590358
// MI455X (gfx1250) — matrix operand order repaired and checked by reading, not compiled, not run on hardware
//
#include <hip/hip_runtime.h>
#include <hip/hip_bf16.h>

typedef _Float16 half_t;
typedef __attribute__((ext_vector_type(16))) _Float16 v16h;
typedef __attribute__((ext_vector_type(8)))  _Float16 v8h;
typedef __attribute__((ext_vector_type(8)))  float    v8f;
typedef __attribute__((ext_vector_type(4)))  unsigned int u32x4;
typedef __attribute__((ext_vector_type(8)))  int          i32x8;
typedef __attribute__((ext_vector_type(4)))  int          i32x4;

__device__ __forceinline__ v8f wmma_f16(v16h a, v16h b, v8f c) {
  // D = A(16x32 f16) * B(32x16 f16) + C(16x16 f32)
  return __builtin_amdgcn_wmma_f32_16x16x32_f16(false, a, false, b, (short)0, c, false, false);
}
__device__ __forceinline__ v8f zero8f() {
  v8f z;
#pragma unroll
  for (int i = 0; i < 8; ++i) z[i] = 0.f;
  return z;
}
__device__ __forceinline__ v16h zero16h() {
  v16h z;
#pragma unroll
  for (int i = 0; i < 16; ++i) z[i] = (half_t)0.f;
  return z;
}

// ---------------------------------------------------------------------------
// Stage a contiguous block of weights (global -> LDS offset 0).
// Preferred path: Tensor Data Mover (tensor_load_to_lds), one descriptor,
// issued by thread 0 only (scalar branch skips it on other waves), completion
// via s_wait_tensorcnt + workgroup barrier. Fallback: cooperative v8h copy.
// This toolchain has the 6-arg builtin: (u32x4, i32x8, i32x4, i32x4, i32x8, i32).
// ---------------------------------------------------------------------------
__device__ __forceinline__ void stage_weights(half_t* wl, const half_t* wgt, int bytes)
{
#if __has_builtin(__builtin_amdgcn_tensor_load_to_lds) && __has_builtin(__builtin_amdgcn_s_wait_tensorcnt)
  if (threadIdx.x == 0) {
    const unsigned long long ga = (unsigned long long)(size_t)wgt;
    const unsigned nd = ((unsigned)bytes) >> 2;           // # 4-byte elements
    u32x4 g0;
    g0[0] = 1u;                                           // count=1 (valid), user mode
    g0[1] = 0u;                                           // lds_addr = 0 (dynamic LDS base)
    g0[2] = (unsigned)(ga & 0xFFFFFFFFu);                 // global_addr[31:0]
    g0[3] = (unsigned)((ga >> 32) & 0x01FFFFFFu) | (2u << 30); // addr[56:32] | type=2
    i32x8 g1;
    g1[0] = (int)(2u << 16);                              // data_size=4B; mask=0; no flags
    g1[1] = (int)((nd & 0xFFFFu) << 16);                  // tensor_dim0[15:0]  @ bits 63:48
    g1[2] = (int)((nd >> 16) & 0xFFFFu);                  // tensor_dim0[31:16]; tensor_dim1=0
    g1[3] = (int)((nd & 0xFFFFu) << 16);                  // tile_dim0 @ bits 127:112
    g1[4] = 0;                                            // tile_dim1=0 (1-D), tile_dim2=0
    g1[5] = (int)nd;                                      // tensor_dim0_stride[31:0]
    g1[6] = 0;                                            // stride hi / dim1_stride lo
    g1[7] = 0;
    i32x4 z4; z4[0] = z4[1] = z4[2] = z4[3] = 0;
    i32x8 z8;
#pragma unroll
    for (int i = 0; i < 8; ++i) z8[i] = 0;
    __builtin_amdgcn_tensor_load_to_lds(g0, g1, z4, z4, z8, 0);
    __builtin_amdgcn_s_wait_tensorcnt(0);
  }
  (void)wl;
  __syncthreads();
#else
  const int tot = bytes >> 1;                             // halfs
  for (int i = threadIdx.x * 8; i < tot; i += 256 * 8)
    *(v8h*)(wl + i) = *(const v8h*)(wgt + i);
  __syncthreads();
#endif
}

// ---------------------------------------------------------------------------
// Implicit-GEMM 3x3 conv (pad=1), NHWC f16 activations, f32 accumulate.
// Wave tile: 32 pixels x COUTP channels (two A tiles share each B tile,
// doubling WMMA issued per ds_load). Optional: nearest-2x upsampled input,
// leaky-relu, residual add, f32 NCHW out. Weights [9][COUTP][CINP] f16 in LDS.
// Block = 256 threads (8 waves); block tile = 8 rows x 32 cols.
// ---------------------------------------------------------------------------
template<int CINP, int COUTP>
__global__ __launch_bounds__(256)
void k_conv3x3(const half_t* __restrict__ in, int Hin, int Win,
               const half_t* __restrict__ wgt, const float* __restrict__ bias,
               int Cout, half_t* __restrict__ out, const half_t* __restrict__ resid,
               float* __restrict__ out_nchw, int Hout, int Wout,
               int upsample, float slope)
{
  constexpr int NKT = CINP / 32;
  constexpr int NNT = COUTP / 16;
  extern __shared__ char smem[];
  half_t* wl = (half_t*)smem;
  stage_weights(wl, wgt, 9 * COUTP * CINP * 2);

  const int lane = threadIdx.x & 31;
  const int wave = threadIdx.x >> 5;
  const int m    = lane & 15;
  const int kh   = lane >> 4;
  const int oy   = blockIdx.y * 8 + wave;
  const int ox0  = blockIdx.x * 32;
  const int px0  = ox0 + m;        // A-tile 0 pixel
  const int px1  = ox0 + 16 + m;   // A-tile 1 pixel

  v8f acc0[NNT], acc1[NNT];
#pragma unroll
  for (int n = 0; n < NNT; ++n) { acc0[n] = zero8f(); acc1[n] = zero8f(); }

  const int limH = upsample ? (Hin << 1) : Hin;
  const int limW = upsample ? (Win << 1) : Win;

#pragma unroll
  for (int dy = 0; dy < 3; ++dy) {
#pragma unroll
    for (int dx = 0; dx < 3; ++dx) {
      const int t  = dy * 3 + dx;
      const int ry = oy + dy - 1;
      const bool vy = (ry >= 0) && (ry < limH);
      const int iy = upsample ? (ry >> 1) : ry;

      const int rx0 = px0 + dx - 1;
      const int rx1 = px1 + dx - 1;
      const bool v0 = vy && (rx0 >= 0) && (rx0 < limW);
      const bool v1 = vy && (rx1 >= 0) && (rx1 < limW);
      const int ix0 = upsample ? (rx0 >> 1) : rx0;
      const int ix1 = upsample ? (rx1 >> 1) : rx1;
      const half_t* prow = in + (size_t)iy * Win * CINP;
#pragma unroll
      for (int kt = 0; kt < NKT; ++kt) {
        v16h a0, a1;
        if (v0) {
          const half_t* pa = prow + (size_t)ix0 * CINP + kt * 32 + kh * 8;
          __builtin_prefetch(pa + (size_t)Win * CINP, 0, 1);
          v8h lo = *(const v8h*)(pa);
          v8h hi = *(const v8h*)(pa + 16);
#pragma unroll
          for (int e = 0; e < 8; ++e) { a0[e] = lo[e]; a0[8 + e] = hi[e]; }
        } else a0 = zero16h();
        if (v1) {
          const half_t* pa = prow + (size_t)ix1 * CINP + kt * 32 + kh * 8;
          v8h lo = *(const v8h*)(pa);
          v8h hi = *(const v8h*)(pa + 16);
#pragma unroll
          for (int e = 0; e < 8; ++e) { a1[e] = lo[e]; a1[8 + e] = hi[e]; }
        } else a1 = zero16h();
#pragma unroll
        for (int n = 0; n < NNT; ++n) {
          const half_t* pb = wl + ((size_t)(t * COUTP + n * 16 + m)) * CINP + kt * 32 + kh * 8;
          v8h blo = *(const v8h*)(pb);
          v8h bhi = *(const v8h*)(pb + 16);
          v16h b;
#pragma unroll
          for (int e = 0; e < 8; ++e) { b[e] = blo[e]; b[8 + e] = bhi[e]; }
          acc0[n] = wmma_f16(a0, b, acc0[n]);
          acc1[n] = wmma_f16(a1, b, acc1[n]);
        }
      }
    }
  }

#pragma unroll
  for (int n = 0; n < NNT; ++n) {
    const int c = n * 16 + m;
    const float bb = (c < Cout) ? bias[c] : 0.f;
#pragma unroll
    for (int r = 0; r < 8; ++r) {
      const int mm = r + 8 * kh;
#pragma unroll
      for (int half_sel = 0; half_sel < 2; ++half_sel) {
        const int opx = ox0 + half_sel * 16 + mm;
        float v = (half_sel ? acc1[n][r] : acc0[n][r]) + bb;
        if (slope >= 0.f) v = (v > 0.f) ? v : v * slope;
        const size_t oidx = ((size_t)oy * Wout + opx) * COUTP + c;
        if (resid) v += (float)resid[oidx];
        if (out) out[oidx] = (half_t)v;
        if (out_nchw && c < Cout)
          out_nchw[((size_t)c * Hout + oy) * Wout + opx] = v;
      }
    }
  }
}

// ---------------------------------------------------------------------------
// 1x1 conv as GEMM over flat pixels. Optional per-input-channel f32 scale
// (folds split-attn), residual add. Weights [COUTP][CINP] f16 (BN folded).
// ---------------------------------------------------------------------------
template<int CINP, int COUTP>
__global__ __launch_bounds__(256)
void k_conv1x1(const half_t* __restrict__ in, int P,
               const half_t* __restrict__ wgt, const float* __restrict__ bias,
               int Cin, int Cout, const float* __restrict__ scale,
               const half_t* __restrict__ resid, half_t* __restrict__ out)
{
  constexpr int NKT = CINP / 32;
  constexpr int NNT = COUTP / 16;
  extern __shared__ char smem[];
  half_t* wl = (half_t*)smem;
  stage_weights(wl, wgt, COUTP * CINP * 2);

  const int lane = threadIdx.x & 31;
  const int wave = threadIdx.x >> 5;
  const int m    = lane & 15;
  const int kh   = lane >> 4;
  const int p0   = (blockIdx.x * 8 + wave) * 16;
  (void)P;

  v8f acc[NNT];
#pragma unroll
  for (int n = 0; n < NNT; ++n) acc[n] = zero8f();

  const half_t* pin = in + (size_t)(p0 + m) * CINP;
#pragma unroll
  for (int kt = 0; kt < NKT; ++kt) {
    v16h a;
    const half_t* pa = pin + kt * 32 + kh * 8;
    v8h lo = *(const v8h*)(pa);
    v8h hi = *(const v8h*)(pa + 16);
#pragma unroll
    for (int e = 0; e < 8; ++e) { a[e] = lo[e]; a[8 + e] = hi[e]; }
    if (scale) {
#pragma unroll
      for (int e = 0; e < 8; ++e) {
        const int k0 = kt * 32 + kh * 8 + e;
        const int k1 = k0 + 16;
        if (k0 < Cin) a[e]     = (half_t)((float)a[e]     * scale[k0]);
        if (k1 < Cin) a[8 + e] = (half_t)((float)a[8 + e] * scale[k1]);
      }
    }
#pragma unroll
    for (int n = 0; n < NNT; ++n) {
      const half_t* pb = wl + ((size_t)(n * 16 + m)) * CINP + kt * 32 + kh * 8;
      v8h blo = *(const v8h*)(pb);
      v8h bhi = *(const v8h*)(pb + 16);
      v16h b;
#pragma unroll
      for (int e = 0; e < 8; ++e) { b[e] = blo[e]; b[8 + e] = bhi[e]; }
      acc[n] = wmma_f16(a, b, acc[n]);
    }
  }

#pragma unroll
  for (int n = 0; n < NNT; ++n) {
    const int c = n * 16 + m;
    const float bb = (c < Cout) ? bias[c] : 0.f;
#pragma unroll
    for (int r = 0; r < 8; ++r) {
      const int pp = p0 + r + 8 * kh;
      const size_t idx = (size_t)pp * COUTP + c;
      float v = acc[n][r] + bb;
      if (resid) v += (float)resid[idx];
      out[idx] = (half_t)v;
    }
  }
}

// ---------------------------------------------------------------------------
// Window attention: S = q @ qT (WMMA, K = channels padded to 32), row softmax
// on a wave-private LDS strip, store attn as f16 [nwin][ws2][ws2].
// One wave per window; 8 windows per block.
// ---------------------------------------------------------------------------
template<int WS>
__global__ __launch_bounds__(256)
void k_attn_qq(const half_t* __restrict__ h, int hp, int qoff, int cg,
               int H, int W, int shift, half_t* __restrict__ atn)
{
  constexpr int WS2 = WS * WS;
  constexpr int NT  = WS2 / 16;
  extern __shared__ char smem[];
  float* strip = (float*)smem + (threadIdx.x >> 5) * (16 * WS2);
  const int lane = threadIdx.x & 31;
  const int m    = lane & 15;
  const int kh   = lane >> 4;
  const int nwx  = W / WS;
  const int win  = blockIdx.x * 8 + (threadIdx.x >> 5);
  const int wy   = win / nwx, wx = win % nwx;

  for (int mt = 0; mt < NT; ++mt) {
    const int pm = mt * 16 + m;
    const int r = pm / WS, c = pm % WS;
    int gy = wy * WS + r + shift; if (gy >= H) gy -= H;
    int gx = wx * WS + c + shift; if (gx >= W) gx -= W;
    const half_t* pa = h + ((size_t)gy * W + gx) * hp + qoff;
    v16h a;
    const int klo = kh * 8;
#pragma unroll
    for (int e = 0; e < 8; ++e) {
      a[e]     = (klo + e      < cg) ? pa[klo + e]      : (half_t)0.f;
      a[8 + e] = (klo + 16 + e < cg) ? pa[klo + 16 + e] : (half_t)0.f;
    }
    for (int nt = 0; nt < NT; ++nt) {
      const int pn = nt * 16 + m;
      const int rn = pn / WS, cn = pn % WS;
      int gyn = wy * WS + rn + shift; if (gyn >= H) gyn -= H;
      int gxn = wx * WS + cn + shift; if (gxn >= W) gxn -= W;
      const half_t* pb = h + ((size_t)gyn * W + gxn) * hp + qoff;
      v16h b;
      const int kb = kh * 8;
#pragma unroll
      for (int e = 0; e < 16; ++e)
        b[e] = (kb + e + (e & 8) < cg) ? pb[kb + e + (e & 8)] : (half_t)0.f;
      v8f s = wmma_f16(a, b, zero8f());
#pragma unroll
      for (int rr = 0; rr < 8; ++rr)
        strip[(rr + 8 * kh) * WS2 + nt * 16 + m] = s[rr];
    }
    // softmax over each of the 16 rows of this strip (wave-private LDS)
    if (lane < 16) {
      float* rp = strip + lane * WS2;
      float mx = -3.0e38f;
      for (int j = 0; j < WS2; ++j) mx = fmaxf(mx, rp[j]);
      float sum = 0.f;
      for (int j = 0; j < WS2; ++j) sum += __expf(rp[j] - mx);
      const float inv = 1.f / sum;
      half_t* op = atn + ((size_t)win * WS2 + (mt * 16 + lane)) * WS2;
      for (int j = 0; j < WS2; ++j) op[j] = (half_t)(__expf(rp[j] - mx) * inv);
    }
  }
}

// ---------------------------------------------------------------------------
// y_win = atn @ v, scattered back to NHWC y buffer (from_windows + roll).
// ---------------------------------------------------------------------------
template<int WS>
__global__ __launch_bounds__(256)
void k_attn_apply(const half_t* __restrict__ h, int hp, int voff, int cg,
                  const half_t* __restrict__ atn, int H, int W, int shift,
                  half_t* __restrict__ y, int yp, int obase)
{
  constexpr int WS2 = WS * WS;
  constexpr int NT  = WS2 / 16;
  constexpr int KT  = (WS2 + 31) / 32;
  const int lane = threadIdx.x & 31;
  const int m    = lane & 15;
  const int kh   = lane >> 4;
  const int nwx  = W / WS;
  const int win  = blockIdx.x * 8 + (threadIdx.x >> 5);
  const int wy   = win / nwx, wx = win % nwx;

  for (int mt = 0; mt < NT; ++mt) {
    v8f acc0 = zero8f(), acc1 = zero8f();
    const int pm = mt * 16 + m;
    const half_t* parow = atn + ((size_t)win * WS2 + pm) * WS2;
    for (int kt = 0; kt < KT; ++kt) {
      v16h a;
      const int klo = kt * 32 + kh * 8;
#pragma unroll
      for (int e = 0; e < 8; ++e) {
        const int k0 = klo + e, k1 = klo + 16 + e;
        a[e]     = (k0 < WS2) ? parow[k0] : (half_t)0.f;
        a[8 + e] = (k1 < WS2) ? parow[k1] : (half_t)0.f;
      }
      const int kb = kt * 32 + kh * 8;
      {
        const int n = m;
        v16h b;
#pragma unroll
        for (int e = 0; e < 16; ++e) {
          const int pk = kb + e + (e & 8);
          half_t val = (half_t)0.f;
          if (n < cg && pk < WS2) {
            const int rk = pk / WS, ck = pk % WS;
            int gy = wy * WS + rk + shift; if (gy >= H) gy -= H;
            int gx = wx * WS + ck + shift; if (gx >= W) gx -= W;
            val = h[((size_t)gy * W + gx) * hp + voff + n];
          }
          b[e] = val;
        }
        acc0 = wmma_f16(a, b, acc0);
      }
      {
        const int n = 16 + m;
        v16h b;
#pragma unroll
        for (int e = 0; e < 16; ++e) {
          const int pk = kb + e + (e & 8);
          half_t val = (half_t)0.f;
          if (n < cg && pk < WS2) {
            const int rk = pk / WS, ck = pk % WS;
            int gy = wy * WS + rk + shift; if (gy >= H) gy -= H;
            int gx = wx * WS + ck + shift; if (gx >= W) gx -= W;
            val = h[((size_t)gy * W + gx) * hp + voff + n];
          }
          b[e] = val;
        }
        acc1 = wmma_f16(a, b, acc1);
      }
    }
#pragma unroll
    for (int rr = 0; rr < 8; ++rr) {
      const int pmm = mt * 16 + rr + 8 * kh;
      const int rk = pmm / WS, ck = pmm % WS;
      int gy = wy * WS + rk + shift; if (gy >= H) gy -= H;
      int gx = wx * WS + ck + shift; if (gx >= W) gx -= W;
      half_t* op = y + ((size_t)gy * W + gx) * yp + obase;
      if (m < cg)      op[m]      = (half_t)acc0[rr];
      if (16 + m < cg) op[16 + m] = (half_t)acc1[rr];
    }
  }
}

// ---------------------------------------------------------------------------
// Global-average-pool of (sum of 3 channel groups); partial sums via atomics.
// ---------------------------------------------------------------------------
__global__ __launch_bounds__(256)
void k_gap(const half_t* __restrict__ y, int P, int yp, int cg, float* __restrict__ gap)
{
  const int j  = threadIdx.x & 31;
  const int ph = threadIdx.x >> 5;
  if (j >= cg) return;
  float s = 0.f;
  int pend = (int)((blockIdx.x + 1) * 1024);
  if (pend > P) pend = P;
  for (int pp = blockIdx.x * 1024 + ph; pp < pend; pp += 8) {
    const half_t* r = y + (size_t)pp * yp;
    s += (float)r[j] + (float)r[cg + j] + (float)r[2 * cg + j];
  }
  atomicAdd(&gap[j], s);
}

// SE MLP + softmax over r=3 -> per-channel scale vector svec[3*cg]
__global__ void k_se(const float* __restrict__ gap_in, float invP, int cg,
                     const float* __restrict__ fc1w, const float* __restrict__ fc1b,
                     const float* __restrict__ fc2w, const float* __restrict__ fc2b,
                     float* __restrict__ svec)
{
  __shared__ float g[32], z[32], l[96];
  const int t = threadIdx.x;
  if (t < cg) g[t] = gap_in[t] * invP;
  __syncthreads();
  if (t < cg) {
    float s = fc1b[t];
    for (int k = 0; k < cg; ++k) s += g[k] * fc1w[k * cg + t];
    z[t] = fmaxf(s, 0.f);
  }
  __syncthreads();
  if (t < 3 * cg) {
    float s = fc2b[t];
    for (int k = 0; k < cg; ++k) s += z[k] * fc2w[k * 3 * cg + t];
    l[t] = s;
  }
  __syncthreads();
  if (t < cg) {
    const float a = __expf(l[t]), b = __expf(l[cg + t]), c = __expf(l[2 * cg + t]);
    const float inv = 1.f / (a + b + c);
    svec[t] = a * inv; svec[cg + t] = b * inv; svec[2 * cg + t] = c * inv;
  }
}

// BN fold: scale = g*rsqrt(v+eps); bias = b*scale + beta - m*scale
__global__ void k_bnfold(const float* __restrict__ g, const float* __restrict__ be,
                         const float* __restrict__ mm, const float* __restrict__ vv,
                         const float* __restrict__ b, int C,
                         float* __restrict__ scale, float* __restrict__ bias)
{
  const int i = threadIdx.x;
  if (i < C) {
    const float inv = g[i] * rsqrtf(vv[i] + 1e-5f);
    scale[i] = inv;
    bias[i]  = b[i] * inv + be[i] - mm[i] * inv;
  }
}

// OIHW f32 -> [9][OP][IP] f16 (zero padded), optional per-O scale
__global__ void k_w3_prep(const float* __restrict__ w, int O, int I, int OP, int IP,
                          const float* __restrict__ scale, half_t* __restrict__ out)
{
  const int idx = blockIdx.x * 256 + threadIdx.x;
  const int tot = 9 * OP * IP;
  if (idx >= tot) return;
  const int ii = idx % IP;
  const int oo = (idx / IP) % OP;
  const int t  = idx / (IP * OP);
  float v = 0.f;
  if (oo < O && ii < I) {
    v = w[((size_t)(oo * I + ii)) * 9 + t];
    if (scale) v *= scale[oo];
  }
  out[idx] = (half_t)v;
}

// OI11 f32 -> [OP][IP] f16 (zero padded), optional per-O scale
__global__ void k_w1_prep(const float* __restrict__ w, int O, int I, int OP, int IP,
                          const float* __restrict__ scale, half_t* __restrict__ out)
{
  const int idx = blockIdx.x * 256 + threadIdx.x;
  const int tot = OP * IP;
  if (idx >= tot) return;
  const int ii = idx % IP;
  const int oo = idx / IP;
  float v = 0.f;
  if (oo < O && ii < I) {
    v = w[(size_t)oo * I + ii];
    if (scale) v *= scale[oo];
  }
  out[idx] = (half_t)v;
}

// x (3,H,W) f32 NCHW -> NHWC f16 padded to 32 channels
__global__ void k_x_prep(const float* __restrict__ x, int H, int W, half_t* __restrict__ out)
{
  const int p = blockIdx.x * 256 + threadIdx.x;
  if (p >= H * W) return;
  half_t* o = out + (size_t)p * 32;
#pragma unroll
  for (int c = 0; c < 32; ++c)
    o[c] = (c < 3) ? (half_t)x[(size_t)c * H * W + p] : (half_t)0.f;
}

// ---------------------------------------------------------------------------
extern "C" void kernel_launch(void* const* d_in, const int* in_sizes, int n_in,
                              void* d_out, int out_size, void* d_ws, size_t ws_size,
                              hipStream_t stream)
{
  (void)in_sizes; (void)n_in; (void)out_size; (void)ws_size;
  const float* x   = (const float*)d_in[0];
  const float* c1w = (const float*)d_in[1];
  const float* c1b = (const float*)d_in[2];
  const float* c2w = (const float*)d_in[51];
  const float* c2b = (const float*)d_in[52];
  const float* upw = (const float*)d_in[53];
  const float* upb = (const float*)d_in[54];
  const float* c3w = (const float*)d_in[55];
  const float* c3b = (const float*)d_in[56];
  const float* c4w = (const float*)d_in[57];
  const float* c4b = (const float*)d_in[58];

  const int H = 480, W = 480;
  const size_t P0 = (size_t)H * W;

  char* p = (char*)d_ws;
  auto carve = [&](size_t b) -> char* {
    char* r = p; p += (b + 255) & ~(size_t)255; return r;
  };
  half_t* xin   = (half_t*)carve(P0 * 32 * 2);
  half_t* out1  = (half_t*)carve(P0 * 64 * 2);
  half_t* hbuf  = (half_t*)carve(P0 * 128 * 2);
  half_t* ybuf  = (half_t*)carve(P0 * 64 * 2);
  half_t* ofin  = (half_t*)carve(P0 * 64 * 2);
  half_t* u1    = (half_t*)carve((size_t)960 * 960 * 64 * 2);
  half_t* u2    = (half_t*)carve((size_t)1920 * 1920 * 64 * 2);
  half_t* u3    = (half_t*)carve((size_t)1920 * 1920 * 64 * 2);
  half_t* atn4  = (half_t*)carve((size_t)14400 * 256 * 2);
  half_t* atn8  = (half_t*)carve((size_t)3600 * 4096 * 2);
  half_t* atn12 = (half_t*)carve((size_t)1600 * 20736 * 2);
  half_t* W1    = (half_t*)carve(9 * 64 * 32 * 2);
  half_t* Wpi   = (half_t*)carve(128 * 64 * 2);
  half_t* Wpo   = (half_t*)carve(64 * 64 * 2);
  half_t* Wc2   = (half_t*)carve(9 * 64 * 64 * 2);
  half_t* Wup   = (half_t*)carve(9 * 64 * 64 * 2);
  half_t* Wc3   = (half_t*)carve(9 * 64 * 64 * 2);
  half_t* Wc4   = (half_t*)carve(9 * 16 * 64 * 2);
  float* bnscale = (float*)carve(128 * 4);
  float* bnbias  = (float*)carve(128 * 4);
  float* gap     = (float*)carve(32 * 4);
  float* svec    = (float*)carve(64 * 4);

  // ---- prep ----
  k_x_prep<<<(int)((P0 + 255) / 256), 256, 0, stream>>>(x, H, W, xin);
  k_w3_prep<<<(9 * 64 * 32 + 255) / 256, 256, 0, stream>>>(c1w, 60, 3, 64, 32, nullptr, W1);
  k_w3_prep<<<(9 * 64 * 64 + 255) / 256, 256, 0, stream>>>(c2w, 60, 60, 64, 64, nullptr, Wc2);
  k_w3_prep<<<(9 * 64 * 64 + 255) / 256, 256, 0, stream>>>(upw, 60, 60, 64, 64, nullptr, Wup);
  k_w3_prep<<<(9 * 64 * 64 + 255) / 256, 256, 0, stream>>>(c3w, 60, 60, 64, 64, nullptr, Wc3);
  k_w3_prep<<<(9 * 16 * 64 + 255) / 256, 256, 0, stream>>>(c4w, 3, 60, 16, 64, nullptr, Wc4);

  // ---- conv1 ----
  k_conv3x3<32, 64><<<dim3(W / 32, H / 8), 256, 9 * 64 * 32 * 2, stream>>>(
      xin, H, W, W1, c1b, 60, out1, nullptr, nullptr, H, W, 0, -1.f);

  // ---- GMSA block ----
  auto gmsa = [&](int base, bool calc, int shift) {
    const float* pi_w = (const float*)d_in[base + 0];
    const float* pi_b = (const float*)d_in[base + 1];
    const float* bn_g = (const float*)d_in[base + 2];
    const float* bn_b = (const float*)d_in[base + 3];
    const float* bn_m = (const float*)d_in[base + 4];
    const float* bn_v = (const float*)d_in[base + 5];
    const float* f1w  = (const float*)d_in[base + 6];
    const float* f1b  = (const float*)d_in[base + 7];
    const float* f2w  = (const float*)d_in[base + 8];
    const float* f2b  = (const float*)d_in[base + 9];
    const float* po_w = (const float*)d_in[base + 10];
    const float* po_b = (const float*)d_in[base + 11];
    const int cin2 = calc ? 120 : 60;
    const int OP   = calc ? 128 : 64;
    const int cs   = cin2 / 3;
    const int cg   = 20;

    k_bnfold<<<1, 128, 0, stream>>>(bn_g, bn_b, bn_m, bn_v, pi_b, cin2, bnscale, bnbias);
    k_w1_prep<<<(OP * 64 + 255) / 256, 256, 0, stream>>>(pi_w, cin2, 60, OP, 64, bnscale, Wpi);
    if (calc)
      k_conv1x1<64, 128><<<(int)(P0 / 128), 256, 128 * 64 * 2, stream>>>(
          out1, (int)P0, Wpi, bnbias, 60, cin2, nullptr, nullptr, hbuf);
    else
      k_conv1x1<64, 64><<<(int)(P0 / 128), 256, 64 * 64 * 2, stream>>>(
          out1, (int)P0, Wpi, bnbias, 60, cin2, nullptr, nullptr, hbuf);

    (void)hipMemsetAsync(ybuf, 0, P0 * 64 * 2, stream);

    { // ws = 4
      const int nw = (H / 4) * (W / 4), s = shift ? 2 : 0;
      const int qo = 0 * cs, vo = calc ? 0 * cs + cg : 0 * cs;
      if (calc) k_attn_qq<4><<<nw / 8, 256, 8 * 16 * 16 * 4, stream>>>(hbuf, OP, qo, cg, H, W, s, atn4);
      k_attn_apply<4><<<nw / 8, 256, 0, stream>>>(hbuf, OP, vo, cg, atn4, H, W, s, ybuf, 64, 0);
    }
    { // ws = 8
      const int nw = (H / 8) * (W / 8), s = shift ? 4 : 0;
      const int qo = 1 * cs, vo = calc ? 1 * cs + cg : 1 * cs;
      if (calc) k_attn_qq<8><<<nw / 8, 256, 8 * 16 * 64 * 4, stream>>>(hbuf, OP, qo, cg, H, W, s, atn8);
      k_attn_apply<8><<<nw / 8, 256, 0, stream>>>(hbuf, OP, vo, cg, atn8, H, W, s, ybuf, 64, 20);
    }
    { // ws = 12
      const int nw = (H / 12) * (W / 12), s = shift ? 6 : 0;
      const int qo = 2 * cs, vo = calc ? 2 * cs + cg : 2 * cs;
      if (calc) k_attn_qq<12><<<nw / 8, 256, 8 * 16 * 144 * 4, stream>>>(hbuf, OP, qo, cg, H, W, s, atn12);
      k_attn_apply<12><<<nw / 8, 256, 0, stream>>>(hbuf, OP, vo, cg, atn12, H, W, s, ybuf, 64, 40);
    }

    (void)hipMemsetAsync(gap, 0, cg * 4, stream);
    k_gap<<<(int)(P0 / 1024), 256, 0, stream>>>(ybuf, (int)P0, 64, cg, gap);
    k_se<<<1, 96, 0, stream>>>(gap, 1.f / (float)P0, cg, f1w, f1b, f2w, f2b, svec);
    k_w1_prep<<<(64 * 64 + 255) / 256, 256, 0, stream>>>(po_w, 60, 60, 64, 64, nullptr, Wpo);
    // out1 += conv1x1(ybuf * svec)
    k_conv1x1<64, 64><<<(int)(P0 / 128), 256, 64 * 64 * 2, stream>>>(
        ybuf, (int)P0, Wpo, po_b, 60, 60, svec, out1, out1);
  };

  gmsa(3, true, 0);
  gmsa(15, false, 0);
  gmsa(27, true, 1);
  gmsa(39, false, 1);

  // ---- conv2 (+residual) ----
  k_conv3x3<64, 64><<<dim3(W / 32, H / 8), 256, 9 * 64 * 64 * 2, stream>>>(
      out1, H, W, Wc2, c2b, 60, ofin, out1, nullptr, H, W, 0, -1.f);
  // ---- up x2 (fused nearest-2x into conv input indexing) ----
  k_conv3x3<64, 64><<<dim3(960 / 32, 960 / 8), 256, 9 * 64 * 64 * 2, stream>>>(
      ofin, H, W, Wup, upb, 60, u1, nullptr, nullptr, 960, 960, 1, 0.2f);
  k_conv3x3<64, 64><<<dim3(1920 / 32, 1920 / 8), 256, 9 * 64 * 64 * 2, stream>>>(
      u1, 960, 960, Wup, upb, 60, u2, nullptr, nullptr, 1920, 1920, 1, 0.2f);
  // ---- conv3 + leaky ----
  k_conv3x3<64, 64><<<dim3(1920 / 32, 1920 / 8), 256, 9 * 64 * 64 * 2, stream>>>(
      u2, 1920, 1920, Wc3, c3b, 60, u3, nullptr, nullptr, 1920, 1920, 0, 0.2f);
  // ---- conv4 -> f32 NCHW output ----
  k_conv3x3<64, 16><<<dim3(1920 / 32, 1920 / 8), 256, 9 * 16 * 64 * 2, stream>>>(
      u3, 1920, 1920, Wc4, c4b, 3, nullptr, nullptr, (float*)d_out, 1920, 1920, 0, -1.f);
}
